// SLSTM_46797963657354
// MI455X (gfx1250) — matrix operand order repaired and checked by reading, not compiled, not run on hardware
//
#include <hip/hip_runtime.h>

typedef __attribute__((ext_vector_type(16))) _Float16 v16h;
typedef __attribute__((ext_vector_type(8)))  _Float16 v8h;
typedef __attribute__((ext_vector_type(8)))  float    v8f;

#define Bb 16
#define Ss 2048
#define Hh 256
#define SP (Ss + 2)   // padded rows per batch (zero row at s=-1 and s=Ss)
#define NSEV 1792     // 7*H
#define KK 1024       // 4*H (before,h,after,word_emb)
#define SCH 16        // S reduction chunks
#define SCL (Ss / SCH)

__device__ __forceinline__ float sigmoidf_(float x) {
    return 1.0f / (1.0f + __expf(-x));
}

__device__ __forceinline__ v16h ld16h(const _Float16* p0, const _Float16* p1) {
    v8h a = *(const v8h*)p0;
    v8h b = *(const v8h*)p1;
    v16h r;
    #pragma unroll
    for (int i = 0; i < 8; ++i) { r[i] = a[i]; r[8 + i] = b[i]; }
    return r;
}

// One K-segment (256 wide, 8 k-steps) of the fused gates GEMM.
// ap: this lane's A row for the segment; bp[7]: per-gate B row pointers
// (already offset by half*16); acc: 7 accumulators.
__device__ __forceinline__ void kseg256(const _Float16* __restrict__ ap, int kbase,
                                        const _Float16* const* bp, v8f* acc, int half) {
    #pragma unroll 2
    for (int kk = 0; kk < 256; kk += 32) {
        v16h afrag = ld16h(ap + kk + half * 16, ap + kk + half * 16 + 8);
        v16h bfr[7];
        #pragma unroll
        for (int j = 0; j < 7; ++j) {
            const _Float16* q = bp[j] + kbase + kk;
            bfr[j] = ld16h(q, q + 8);
        }
        #pragma unroll
        for (int j = 0; j < 7; ++j)
            acc[j] = __builtin_amdgcn_wmma_f32_16x16x32_f16(
                false, afrag, false, bfr[j], (short)0, acc[j], false, false);
    }
}

// ---------------------------------------------------------------------------
// Prep: pack+transpose weights to f16:
//   WreT[n][k] (n<1792, k<1024): k<768 -> sW[k][n], else sU[k-768][n]
//   guwT[n][k] (256x256)
// ---------------------------------------------------------------------------
__global__ void prep_kernel(const float* __restrict__ sW, const float* __restrict__ sU,
                            const float* __restrict__ guw,
                            _Float16* __restrict__ WreT, _Float16* __restrict__ guwT) {
    int idx = blockIdx.x * blockDim.x + threadIdx.x;
    const int NW = NSEV * KK;
    if (idx < NW) {
        int n = idx >> 10, k = idx & 1023;
        float v = (k < 768) ? sW[(size_t)k * NSEV + n] : sU[(size_t)(k - 768) * NSEV + n];
        WreT[idx] = (_Float16)v;
    } else if (idx < NW + Hh * Hh) {
        int r = idx - NW;
        int n = r >> 8, k = r & 255;
        guwT[r] = (_Float16)guw[(size_t)k * Hh + n];
    }
}

// ---------------------------------------------------------------------------
// Zero the sentinel rows (s=-1, s=Ss) of both padded h buffers.
// ---------------------------------------------------------------------------
__global__ void zero_hpad_kernel(_Float16* __restrict__ hA, _Float16* __restrict__ hB) {
    int i = blockIdx.x * blockDim.x + threadIdx.x;  // Bb*2*Hh = 8192
    if (i >= Bb * 2 * Hh) return;
    int hh = i & 255;
    int r = i >> 8;                 // 0..31
    int b = r >> 1;
    size_t row = (size_t)b * SP + ((r & 1) ? (SP - 1) : 0);
    hA[row * Hh + hh] = (_Float16)0.0f;
    hB[row * Hh + hh] = (_Float16)0.0f;
}

// ---------------------------------------------------------------------------
// Init: embedding gather + mask; wemb f16 (flat), h0 f16 (padded, masked),
// c0 f32 (flat, masked)
// ---------------------------------------------------------------------------
__global__ void init_kernel(const int* __restrict__ word, const float* __restrict__ mask,
                            const float* __restrict__ emb,
                            _Float16* __restrict__ wemb, _Float16* __restrict__ h0,
                            float* __restrict__ c0) {
    size_t idx = (size_t)blockIdx.x * blockDim.x + threadIdx.x;
    if (idx >= (size_t)Bb * Ss * Hh) return;
    int hh = (int)(idx & 255);
    size_t bs = idx >> 8;
    int b = (int)(bs >> 11), s = (int)(bs & (Ss - 1));
    int w = word[bs];
    float m = mask[bs];
    float e = emb[(size_t)w * Hh + hh];
    wemb[idx] = (_Float16)e;
    float hm = e * m;
    h0[((size_t)b * SP + 1 + s) * Hh + hh] = (_Float16)hm;
    c0[idx] = hm;
}

// ---------------------------------------------------------------------------
// Two-stage mean over S. Stage 1: partial sums over S-chunks (coalesced).
// grid = Bb*SCH blocks, 256 threads (one per hidden index).
// ---------------------------------------------------------------------------
__global__ void mean_part_f32_kernel(const float* __restrict__ x, float* __restrict__ part) {
    int blk = blockIdx.x;
    int b = blk >> 4, sc = blk & (SCH - 1);
    int hh = threadIdx.x;
    float s = 0.0f;
    const float* p = x + ((size_t)b * Ss + sc * SCL) * Hh + hh;
    for (int t = 0; t < SCL; ++t) s += p[(size_t)t * Hh];
    part[(size_t)blk * Hh + hh] = s;
}

// padded-h variant
__global__ void mean_part_f16_kernel(const _Float16* __restrict__ x, float* __restrict__ part) {
    int blk = blockIdx.x;
    int b = blk >> 4, sc = blk & (SCH - 1);
    int hh = threadIdx.x;
    float s = 0.0f;
    const _Float16* p = x + ((size_t)b * SP + 1 + sc * SCL) * Hh + hh;
    for (int t = 0; t < SCL; ++t) s += (float)p[(size_t)t * Hh];
    part[(size_t)blk * Hh + hh] = s;
}

__global__ void mean_fin_kernel(const float* __restrict__ part, float* __restrict__ o1,
                                float* __restrict__ o2) {
    int i = blockIdx.x * blockDim.x + threadIdx.x;  // 4096 = B*H
    if (i >= Bb * Hh) return;
    int b = i >> 8, hh = i & 255;
    float s = 0.0f;
    #pragma unroll
    for (int c = 0; c < SCH; ++c) s += part[(size_t)(b * SCH + c) * Hh + hh];
    s *= (1.0f / (float)Ss);
    o1[i] = s;
    if (o2) o2[i] = s;
}

// ---------------------------------------------------------------------------
// sVt[b][n] = sVb[n] + sum_k g[b][k]*sVw[k][n]    (16 x 1792)
// ---------------------------------------------------------------------------
__global__ void svt_kernel(const float* __restrict__ g, const float* __restrict__ sVw,
                           const float* __restrict__ sVb, float* __restrict__ sVt) {
    int i = blockIdx.x * blockDim.x + threadIdx.x;  // 28672
    if (i >= Bb * NSEV) return;
    int b = i / NSEV, n = i % NSEV;
    float acc = sVb[n];
    for (int k = 0; k < Hh; ++k) acc += g[b * Hh + k] * sVw[(size_t)k * NSEV + n];
    sVt[i] = acc;
}

// ---------------------------------------------------------------------------
// fo[b][n] = sigmoid(g@gW + havg@gUw + gUb), n<512 ; gvec[b][h] = g@gw
// ---------------------------------------------------------------------------
__global__ void gcell_small_kernel(const float* __restrict__ g, const float* __restrict__ havg,
                                   const float* __restrict__ gW, const float* __restrict__ gw_,
                                   const float* __restrict__ gUw, const float* __restrict__ gUb,
                                   float* __restrict__ gvec, float* __restrict__ fo) {
    int i = blockIdx.x * blockDim.x + threadIdx.x;  // 8192
    if (i >= Bb * 2 * Hh) return;
    int b = i >> 9, n = i & 511;
    float a = gUb[n];
    for (int k = 0; k < Hh; ++k)
        a += g[b * Hh + k] * gW[(size_t)k * 512 + n] + havg[b * Hh + k] * gUw[(size_t)k * 512 + n];
    fo[i] = sigmoidf_(a);
    if (n < Hh) {
        float v = 0.0f;
        for (int k = 0; k < Hh; ++k) v += g[b * Hh + k] * gw_[(size_t)k * Hh + n];
        gvec[b * Hh + n] = v;
    }
}

// ---------------------------------------------------------------------------
// Main fused SCell kernel: gates GEMM (M=32768, N=1792, K=1024) via WMMA,
// with the 7-gate softmax cell update fused into the epilogue.
// h is padded per batch (zero rows at s=-1, s=Ss): the before/h/after
// K-segments are three unconditional global load streams. The four segments
// are inlined (no pointer-array phi) so address-space inference keeps all
// loads on the global path.
// grid: (256 m-blocks of 128 rows, 16 hidden chunks); block 256 = 8 waves.
// ---------------------------------------------------------------------------
__global__ __launch_bounds__(256)
void sgates_kernel(const _Float16* __restrict__ h_old, const _Float16* __restrict__ wemb,
                   const _Float16* __restrict__ WreT, const float* __restrict__ sVt,
                   const float* __restrict__ c_old, const float* __restrict__ cg,
                   const float* __restrict__ mask,
                   _Float16* __restrict__ h_new, float* __restrict__ c_new) {
    const int lane = threadIdx.x & 31;
    const int wave = threadIdx.x >> 5;
    const int half = lane >> 4;
    const int lm = lane & 15;

    const int m0 = blockIdx.x * 128 + wave * 16;
    const int hc = blockIdx.y;              // hidden chunk 0..15

    // ---- A-row (this lane carries M = m0+lm during the K loop) ----
    const int arow = m0 + lm;
    const int ab = arow >> 11;
    const int as = arow & (Ss - 1);
    const _Float16* hbase = h_old + ((size_t)ab * SP + 1 + as) * Hh;

    v8f acc[7];
    #pragma unroll
    for (int j = 0; j < 7; ++j)
        #pragma unroll
        for (int i = 0; i < 8; ++i) acc[j][i] = 0.0f;

    const _Float16* bp[7];
    #pragma unroll
    for (int j = 0; j < 7; ++j)
        bp[j] = WreT + (size_t)(j * Hh + hc * 16 + lm) * KK + half * 16;

    kseg256(hbase - Hh, 0,   bp, acc, half);          // before (zero pad at s=-1)
    kseg256(hbase,      256, bp, acc, half);          // h
    kseg256(hbase + Hh, 512, bp, acc, half);          // after  (zero pad at s=Ss)
    kseg256(wemb + (size_t)arow * Hh, 768, bp, acc, half);  // word_emb

    // ---- Epilogue: lane now carries N = hc*16+lm, rows m0+half*8+i ----
    const int hh = hc * 16 + lm;
    const int tb = m0 >> 11;                       // batch (tile never crosses batch)
    const int s_base = (m0 & (Ss - 1)) + half * 8;

    float svt[7];
    #pragma unroll
    for (int j = 0; j < 7; ++j) svt[j] = sVt[tb * NSEV + j * Hh + hh];
    const float cgv = cg[tb * Hh + hh];

    #pragma unroll
    for (int i = 0; i < 8; ++i) {
        const int ss = s_base + i;
        const size_t rowbs = (size_t)tb * Ss + ss;
        float gi = acc[0][i] + svt[0];
        float gl = acc[1][i] + svt[1];
        float gr = acc[2][i] + svt[2];
        float gf = acc[3][i] + svt[3];
        float gs = acc[4][i] + svt[4];
        float go = acc[5][i] + svt[5];
        float gu = acc[6][i] + svt[6];
        // softmax over sigmoid(i,l,r,f,s); values in (0,1) -> exp safe
        float ei = __expf(sigmoidf_(gi));
        float el = __expf(sigmoidf_(gl));
        float er = __expf(sigmoidf_(gr));
        float ef = __expf(sigmoidf_(gf));
        float es = __expf(sigmoidf_(gs));
        float inv = 1.0f / (ei + el + er + ef + es);
        float u = tanhf(gu);
        float cb = (ss > 0)      ? c_old[(rowbs - 1) * Hh + hh] : 0.0f;
        float cm = c_old[rowbs * Hh + hh];
        float ca = (ss < Ss - 1) ? c_old[(rowbs + 1) * Hh + hh] : 0.0f;
        float nc = cb * (el * inv) + cm * (ef * inv) + (er * inv) * ca +
                   (es * inv) * cgv + (ei * inv) * u;
        float nh = go * tanhf(nc);
        float mm = mask[rowbs];
        c_new[rowbs * Hh + hh] = nc * mm;
        h_new[((size_t)tb * SP + 1 + ss) * Hh + hh] = (_Float16)(nh * mm);
    }
}

// ---------------------------------------------------------------------------
// t = h_old @ guw : M=32768, N=256, K=256 via WMMA (per-wave 16x16 tile)
// grid: (256 m-blocks, 16 n-tiles), block 256 = 8 waves. h_old is padded.
// ---------------------------------------------------------------------------
__global__ __launch_bounds__(256)
void guw_gemm_kernel(const _Float16* __restrict__ h_old, const _Float16* __restrict__ guwT,
                     float* __restrict__ t) {
    const int lane = threadIdx.x & 31;
    const int wave = threadIdx.x >> 5;
    const int half = lane >> 4;
    const int lm = lane & 15;

    const int m0 = blockIdx.x * 128 + wave * 16;
    const int nt = blockIdx.y;

    const int arow = m0 + lm;
    const int ab = arow >> 11;
    const int as = arow & (Ss - 1);
    const _Float16* ap = h_old + ((size_t)ab * SP + 1 + as) * Hh;
    const _Float16* bp = guwT + (size_t)(nt * 16 + lm) * Hh + half * 16;

    v8f acc;
    #pragma unroll
    for (int i = 0; i < 8; ++i) acc[i] = 0.0f;

    v16h afr[8], bfr[8];
    #pragma unroll
    for (int kb = 0; kb < Hh / 32; ++kb) {
        const int k0 = kb * 32;
        afr[kb] = ld16h(ap + k0 + half * 16, ap + k0 + half * 16 + 8);
        bfr[kb] = ld16h(bp + k0, bp + k0 + 8);
    }
    #pragma unroll
    for (int kb = 0; kb < Hh / 32; ++kb)
        acc = __builtin_amdgcn_wmma_f32_16x16x32_f16(
            false, afr[kb], false, bfr[kb], (short)0, acc, false, false);

    const int n = nt * 16 + lm;
    #pragma unroll
    for (int i = 0; i < 8; ++i)
        t[(size_t)(m0 + half * 8 + i) * Hh + n] = acc[i];
}

// ---------------------------------------------------------------------------
// f_w softmax over S + weighted c-sum, two-stage.
// Stage 1: partial den = sum exp(p), partial num = sum c*exp(p) per S-chunk.
// grid = Bb*SCH blocks, 256 threads (one per hidden index). Coalesced.
// ---------------------------------------------------------------------------
__global__ void fw_part_kernel(const float* __restrict__ t, const float* __restrict__ gvec,
                               const float* __restrict__ gub, const float* __restrict__ c_old,
                               const float* __restrict__ mask,
                               float* __restrict__ pden, float* __restrict__ pnum) {
    int blk = blockIdx.x;
    int b = blk >> 4, sc = blk & (SCH - 1);
    int hh = threadIdx.x;
    float base = gvec[b * Hh + hh] + gub[hh];
    float den = 0.0f, num = 0.0f;
    const int s0 = sc * SCL;
    for (int s = s0; s < s0 + SCL; ++s) {
        size_t off = ((size_t)b * Ss + s) * Hh + hh;
        float p = sigmoidf_(base + t[off]) - (1.0f - mask[b * Ss + s]) * 1e-25f;
        float e = __expf(p);
        den += e;
        num += c_old[off] * e;
    }
    pden[(size_t)blk * Hh + hh] = den;
    pnum[(size_t)blk * Hh + hh] = num;
}

__global__ void fw_fin_kernel(const float* __restrict__ pden, const float* __restrict__ pnum,
                              float* __restrict__ wsum) {
    int i = blockIdx.x * blockDim.x + threadIdx.x;  // 4096
    if (i >= Bb * Hh) return;
    int b = i >> 8, hh = i & 255;
    float d = 0.0f, n = 0.0f;
    #pragma unroll
    for (int c = 0; c < SCH; ++c) {
        d += pden[(size_t)(b * SCH + c) * Hh + hh];
        n += pnum[(size_t)(b * SCH + c) * Hh + hh];
    }
    wsum[i] = n / d;
}

// ---------------------------------------------------------------------------
// cg = fg*cg + wsum ; g = og*tanh(cg)
// ---------------------------------------------------------------------------
__global__ void gupd_kernel(const float* __restrict__ fo, const float* __restrict__ wsum,
                            float* __restrict__ cg, float* __restrict__ g) {
    int i = blockIdx.x * blockDim.x + threadIdx.x;
    if (i >= Bb * Hh) return;
    int b = i >> 8, hh = i & 255;
    float ncg = fo[b * 512 + hh] * cg[i] + wsum[i];
    cg[i] = ncg;
    g[i] = fo[b * 512 + Hh + hh] * tanhf(ncg);
}

// ---------------------------------------------------------------------------
// out[b][l] = g[b]@wout_w[:,l] + wout_b[l]
// ---------------------------------------------------------------------------
__global__ void out_kernel(const float* __restrict__ g, const float* __restrict__ w,
                           const float* __restrict__ bias, float* __restrict__ out) {
    int i = threadIdx.x;
    if (i >= Bb * 5) return;
    int b = i / 5, l = i % 5;
    float a = bias[l];
    for (int k = 0; k < Hh; ++k) a += g[b * Hh + k] * w[k * 5 + l];
    out[i] = a;
}

// ---------------------------------------------------------------------------
extern "C" void kernel_launch(void* const* d_in, const int* in_sizes, int n_in,
                              void* d_out, int out_size, void* d_ws, size_t ws_size,
                              hipStream_t stream) {
    const int*   word  = (const int*)  d_in[0];
    const float* mask  = (const float*)d_in[1];
    const float* emb   = (const float*)d_in[2];
    const float* sW    = (const float*)d_in[3];
    const float* sU    = (const float*)d_in[4];
    const float* sVw   = (const float*)d_in[5];
    const float* sVb   = (const float*)d_in[6];
    const float* gW    = (const float*)d_in[7];
    const float* gw_   = (const float*)d_in[8];
    const float* gUw   = (const float*)d_in[9];
    const float* gUb   = (const float*)d_in[10];
    const float* guw   = (const float*)d_in[11];
    const float* gub   = (const float*)d_in[12];
    const float* woutw = (const float*)d_in[13];
    const float* woutb = (const float*)d_in[14];
    float* out = (float*)d_out;

    char* p = (char*)d_ws;
    auto carve = [&](size_t bytes) -> char* {
        char* r = p;
        p += (bytes + 255) & ~(size_t)255;
        return r;
    };
    const size_t BSH  = (size_t)Bb * Ss * Hh;
    const size_t BSHP = (size_t)Bb * SP * Hh;

    _Float16* WreT = (_Float16*)carve((size_t)NSEV * KK * 2);
    _Float16* guwT = (_Float16*)carve((size_t)Hh * Hh * 2);
    _Float16* wemb = (_Float16*)carve(BSH * 2);
    _Float16* hA   = (_Float16*)carve(BSHP * 2);
    _Float16* hB   = (_Float16*)carve(BSHP * 2);
    float*    cA   = (float*)   carve(BSH * 4);
    float*    cB   = (float*)   carve(BSH * 4);
    float*    tbuf = (float*)   carve(BSH * 4);
    float*    gbuf = (float*)   carve((size_t)Bb * Hh * 4);
    float*    cgb  = (float*)   carve((size_t)Bb * Hh * 4);
    float*    havg = (float*)   carve((size_t)Bb * Hh * 4);
    float*    sVt  = (float*)   carve((size_t)Bb * NSEV * 4);
    float*    gvec = (float*)   carve((size_t)Bb * Hh * 4);
    float*    fo   = (float*)   carve((size_t)Bb * 2 * Hh * 4);
    float*    wsum = (float*)   carve((size_t)Bb * Hh * 4);
    float*    part = (float*)   carve((size_t)Bb * SCH * Hh * 4);
    float*    pden = (float*)   carve((size_t)Bb * SCH * Hh * 4);
    float*    pnum = (float*)   carve((size_t)Bb * SCH * Hh * 4);

    // --- prep + init ---
    {
        int total = NSEV * KK + Hh * Hh;
        prep_kernel<<<(total + 255) / 256, 256, 0, stream>>>(sW, sU, guw, WreT, guwT);
    }
    zero_hpad_kernel<<<(Bb * 2 * Hh + 255) / 256, 256, 0, stream>>>(hA, hB);
    init_kernel<<<(int)((BSH + 255) / 256), 256, 0, stream>>>(word, mask, emb, wemb, hA, cA);
    mean_part_f32_kernel<<<Bb * SCH, 256, 0, stream>>>(cA, part);
    mean_fin_kernel<<<(Bb * Hh + 255) / 256, 256, 0, stream>>>(part, gbuf, cgb);

    _Float16* h_old = hA; _Float16* h_new = hB;
    float* c_old = cA; float* c_new = cB;

    for (int layer = 0; layer < 4; ++layer) {
        mean_part_f16_kernel<<<Bb * SCH, 256, 0, stream>>>(h_old, part);
        mean_fin_kernel<<<(Bb * Hh + 255) / 256, 256, 0, stream>>>(part, havg, (float*)0);
        svt_kernel<<<(Bb * NSEV + 255) / 256, 256, 0, stream>>>(gbuf, sVw, sVb, sVt);
        gcell_small_kernel<<<(Bb * 2 * Hh + 255) / 256, 256, 0, stream>>>(
            gbuf, havg, gW, gw_, gUw, gUb, gvec, fo);

        dim3 grid5(Bb * Ss / 128, 16);
        sgates_kernel<<<grid5, 256, 0, stream>>>(h_old, wemb, WreT, sVt, c_old, cgb, mask,
                                                 h_new, c_new);

        dim3 grid6(Bb * Ss / 128, Hh / 16);
        guw_gemm_kernel<<<grid6, 256, 0, stream>>>(h_old, guwT, tbuf);

        fw_part_kernel<<<Bb * SCH, 256, 0, stream>>>(tbuf, gvec, gub, c_old, mask, pden, pnum);
        fw_fin_kernel<<<(Bb * Hh + 255) / 256, 256, 0, stream>>>(pden, pnum, wsum);
        gupd_kernel<<<(Bb * Hh + 255) / 256, 256, 0, stream>>>(fo, wsum, cgb, gbuf);

        _Float16* th = h_old; h_old = h_new; h_new = th;
        float* tc = c_old; c_old = c_new; c_new = tc;
    }

    out_kernel<<<1, 128, 0, stream>>>(gbuf, woutw, woutb, out);
}
